// EdgeConvBlock_43696997269579
// MI455X (gfx1250) — matrix operand order repaired and checked by reading, not compiled, not run on hardware
//
#include <hip/hip_runtime.h>
#include <hip/hip_bf16.h>

typedef __attribute__((ext_vector_type(16))) __bf16 v16bf;
typedef __attribute__((ext_vector_type(8)))  __bf16 v8bf;
typedef __attribute__((ext_vector_type(8)))  float  v8f;

#define NPTS 1024
#define KNN  16
#define CIN  64
#define COUT 128

// ---------------------------------------------------------------------------
// Small utility kernels
// ---------------------------------------------------------------------------
__global__ void ec_zero_kernel(float* p, int n) {
    int t = blockIdx.x * blockDim.x + threadIdx.x;
    if (t < n) p[t] = 0.0f;
}

__global__ void ec_cvt_w_kernel(const float* __restrict__ w, __bf16* __restrict__ wb, int n) {
    int t = blockIdx.x * blockDim.x + threadIdx.x;
    if (t < n) wb[t] = (__bf16)w[t];
}

// ---------------------------------------------------------------------------
// kNN: per (b,n) thread keeps top-17 (self included, then dropped).
// dist = -(||pn - pm||^2); larger == closer; stable tie-break on index.
// ---------------------------------------------------------------------------
__global__ void ec_knn_kernel(const float* __restrict__ pts, int* __restrict__ idxOut) {
    __shared__ float sx[NPTS], sy[NPTS], sz[NPTS];
    const int b = blockIdx.x >> 2;                    // 4 blocks per batch
    const int n = ((blockIdx.x & 3) << 8) + threadIdx.x;
    const float* pb = pts + (size_t)b * 3 * NPTS;
    for (int i = threadIdx.x; i < NPTS; i += 256) {
        sx[i] = pb[i];
        sy[i] = pb[NPTS + i];
        sz[i] = pb[2 * NPTS + i];
    }
    __syncthreads();
    const float px = sx[n], py = sy[n], pz = sz[n];
    float bestD[KNN + 1];
    int   bestI[KNN + 1];
#pragma unroll
    for (int j = 0; j <= KNN; ++j) { bestD[j] = -3.0e38f; bestI[j] = 0; }
    for (int m = 0; m < NPTS; ++m) {
        const float dx = sx[m] - px, dy = sy[m] - py, dz = sz[m] - pz;
        const float d = -(dx * dx + dy * dy + dz * dz);
        if (d > bestD[KNN]) {
            int j = KNN;
            while (j > 0 && bestD[j - 1] < d) {
                bestD[j] = bestD[j - 1];
                bestI[j] = bestI[j - 1];
                --j;
            }
            bestD[j] = d;
            bestI[j] = m;
        }
    }
    int* o = idxOut + ((size_t)b * NPTS + n) * KNN;
#pragma unroll
    for (int j = 0; j < KNN; ++j) o[j] = bestI[j + 1];   // drop self
}

// ---------------------------------------------------------------------------
// Build x0 (M x 128 bf16): [center(64) | neigh-center(64)], row = ((b*N)+n)*K+k
// ---------------------------------------------------------------------------
__global__ void ec_build_x0_kernel(const float* __restrict__ feat,
                                   const int* __restrict__ idx,
                                   __bf16* __restrict__ x0) {
    const size_t t = (size_t)blockIdx.x * blockDim.x + threadIdx.x;
    const int    c   = (int)(t & 127);
    const size_t row = t >> 7;
    const int    k   = (int)(row & (KNN - 1));
    const size_t bn  = row >> 4;
    const int    n   = (int)(bn & (NPTS - 1));
    const int    b   = (int)(bn >> 10);
    const float* fb  = feat + (size_t)b * CIN * NPTS;
    float v;
    if (c < CIN) {
        v = fb[(size_t)c * NPTS + n];
    } else {
        const int nb = idx[bn * KNN + k];
        const int cc = c - CIN;
        v = fb[(size_t)cc * NPTS + nb] - fb[(size_t)cc * NPTS + n];
    }
    x0[t] = (__bf16)v;
}

// ---------------------------------------------------------------------------
// Build transposed features fT (B*N x 64 bf16) for the shortcut GEMM
// ---------------------------------------------------------------------------
__global__ void ec_build_ft_kernel(const float* __restrict__ feat, __bf16* __restrict__ fT) {
    const size_t t  = (size_t)blockIdx.x * blockDim.x + threadIdx.x;
    const int    c  = (int)(t & 63);
    const size_t bn = t >> 6;
    const int    n  = (int)(bn & (NPTS - 1));
    const int    b  = (int)(bn >> 10);
    fT[t] = (__bf16)feat[(size_t)b * CIN * NPTS + (size_t)c * NPTS + n];
}

// ---------------------------------------------------------------------------
// A-fragment load (16x32 bf16, ISA layout), with optional fused BN+ReLU on the
// input channels (k dimension of this GEMM == channel dim of previous layer).
// ---------------------------------------------------------------------------
__device__ inline v16bf ec_load_a_frag(const __bf16* __restrict__ X,
                                       const float* __restrict__ scale,
                                       const float* __restrict__ shift,
                                       size_t row, int kb, int half, int ldx) {
    const __bf16* p = X + row * (size_t)ldx + kb;
    const v8bf lo = *(const v8bf*)(p + 8 * half);
    const v8bf hi = *(const v8bf*)(p + 16 + 8 * half);
    v16bf a;
    if (scale) {
#pragma unroll
        for (int i = 0; i < 8; ++i) {
            const int k0 = kb + 8 * half + i;
            const int k1 = kb + 16 + 8 * half + i;
            const float x0 = fmaxf((float)lo[i] * scale[k0] + shift[k0], 0.0f);
            const float x1 = fmaxf((float)hi[i] * scale[k1] + shift[k1], 0.0f);
            a[i]     = (__bf16)x0;
            a[8 + i] = (__bf16)x1;
        }
    } else {
#pragma unroll
        for (int i = 0; i < 8; ++i) { a[i] = lo[i]; a[8 + i] = hi[i]; }
    }
    return a;
}

// ---------------------------------------------------------------------------
// WMMA GEMM: Y(M x 128, bf16, pre-BN) = act(X) @ W^T
//   X: M x KDIM bf16 row-major (act = BN+ReLU with scale/shift, or identity)
//   Wb: 128 x KDIM bf16 row-major (B[k][n] = Wb[n*KDIM + k])
// Block: 256 thr (8 waves). Wave w -> cols [16w,16w+16), rows [32*blk, +32).
// ---------------------------------------------------------------------------
template <int KDIM>
__global__ __launch_bounds__(256)
void ec_gemm_kernel(const __bf16* __restrict__ X,
                    const __bf16* __restrict__ Wb,
                    const float* __restrict__ scale,
                    const float* __restrict__ shift,
                    __bf16* __restrict__ Y) {
    const int lane = threadIdx.x & 31;
    const int wave = threadIdx.x >> 5;
    const int half = lane >> 4;
    const int l16  = lane & 15;
    const size_t rowBase = (size_t)blockIdx.x * 32;
    const int    colBase = wave * 16;

    v8f acc0 = {};
    v8f acc1 = {};

#pragma unroll
    for (int kb = 0; kb < KDIM; kb += 32) {
        // B fragment: col = colBase + l16 (fixed per lane), K = kb + 8*half + e (e<8), kb + 16 + 8*half + (e-8) (e>=8)
        v16bf bf;
        {
            const __bf16* bp = Wb + (size_t)(colBase + l16) * KDIM + kb + 8 * half;
            const v8bf b0 = *(const v8bf*)bp;
            const v8bf b1 = *(const v8bf*)(bp + 16);
#pragma unroll
            for (int i = 0; i < 8; ++i) { bf[i] = b0[i]; bf[8 + i] = b1[i]; }
        }
        const v16bf a0 = ec_load_a_frag(X, scale, shift, rowBase + l16,      kb, half, KDIM);
        const v16bf a1 = ec_load_a_frag(X, scale, shift, rowBase + 16 + l16, kb, half, KDIM);
        acc0 = __builtin_amdgcn_wmma_f32_16x16x32_bf16(false, a0, false, bf, (short)0, acc0, false, false);
        acc1 = __builtin_amdgcn_wmma_f32_16x16x32_bf16(false, a1, false, bf, (short)0, acc1, false, false);
    }

    // C/D layout: M = r + 8*half, N = l16
    const int n = colBase + l16;
#pragma unroll
    for (int r = 0; r < 8; ++r) {
        const size_t m0 = rowBase + r + 8 * half;
        const size_t m1 = m0 + 16;
        Y[m0 * COUT + n] = (__bf16)acc0[r];
        Y[m1 * COUT + n] = (__bf16)acc1[r];
    }
}

// ---------------------------------------------------------------------------
// Per-channel sum / sumsq over M rows of a (M x 128) bf16 tensor.
// sums[0:128] = sum, sums[128:256] = sumsq (must be pre-zeroed).
// ---------------------------------------------------------------------------
__global__ void ec_stats_kernel(const __bf16* __restrict__ y, long Mrows,
                                float* __restrict__ sums) {
    __shared__ float ls[COUT], lq[COUT];
    const int t   = threadIdx.x;       // 256
    const int c   = t & 127;
    const int sub = t >> 7;            // 0/1
    if (t < COUT) { ls[t] = 0.0f; lq[t] = 0.0f; }
    __syncthreads();
    float s = 0.0f, q = 0.0f;
    for (long r = (long)blockIdx.x * 2 + sub; r < Mrows; r += (long)gridDim.x * 2) {
        const float v = (float)y[(size_t)r * COUT + c];
        s += v;
        q += v * v;
    }
    atomicAdd(&ls[c], s);
    atomicAdd(&lq[c], q);
    __syncthreads();
    if (t < COUT) {
        atomicAdd(&sums[t], ls[t]);
        atomicAdd(&sums[COUT + t], lq[t]);
    }
}

__global__ void ec_finalize_stats_kernel(const float* __restrict__ sums,
                                         const float* __restrict__ gamma,
                                         const float* __restrict__ beta,
                                         float invCount,
                                         float* __restrict__ scale,
                                         float* __restrict__ shift) {
    const int c = threadIdx.x;  // 128
    const float mean = sums[c] * invCount;
    const float var  = sums[COUT + c] * invCount - mean * mean;
    const float s    = gamma[c] * rsqrtf(var + 1e-5f);
    scale[c] = s;
    shift[c] = beta[c] - mean * s;
}

// ---------------------------------------------------------------------------
// Final combine: out[b][o][n] = relu( bn_sc(sc) + mean_k relu(bn2(y2)) )
// ---------------------------------------------------------------------------
__global__ void ec_combine_kernel(const __bf16* __restrict__ y2,
                                  const __bf16* __restrict__ sc,
                                  const float* __restrict__ s2,
                                  const float* __restrict__ sh2,
                                  const float* __restrict__ ssc,
                                  const float* __restrict__ shsc,
                                  float* __restrict__ out) {
    const size_t t  = (size_t)blockIdx.x * blockDim.x + threadIdx.x; // (b,o,n), n fastest
    const int    n  = (int)(t & (NPTS - 1));
    const size_t bo = t >> 10;
    const int    o  = (int)(bo & 127);
    const int    b  = (int)(bo >> 7);
    const size_t bn = (size_t)b * NPTS + n;
    const float sv2 = s2[o], sh2v = sh2[o];
    float acc = 0.0f;
#pragma unroll
    for (int k = 0; k < KNN; ++k) {
        const float v = (float)y2[(bn * KNN + k) * COUT + o];
        acc += fmaxf(v * sv2 + sh2v, 0.0f);
    }
    acc *= (1.0f / (float)KNN);
    const float scv = (float)sc[bn * COUT + o] * ssc[o] + shsc[o];
    out[t] = fmaxf(scv + acc, 0.0f);
}

// ---------------------------------------------------------------------------
// Launch
// ---------------------------------------------------------------------------
extern "C" void kernel_launch(void* const* d_in, const int* in_sizes, int n_in,
                              void* d_out, int out_size, void* d_ws, size_t ws_size,
                              hipStream_t stream) {
    (void)in_sizes; (void)n_in; (void)out_size; (void)ws_size;

    const float* points   = (const float*)d_in[0];
    const float* features = (const float*)d_in[1];
    const float* W0  = (const float*)d_in[2];
    const float* g0  = (const float*)d_in[3];
    const float* b0  = (const float*)d_in[4];
    const float* W1  = (const float*)d_in[5];
    const float* g1  = (const float*)d_in[6];
    const float* b1  = (const float*)d_in[7];
    const float* W2  = (const float*)d_in[8];
    const float* g2  = (const float*)d_in[9];
    const float* b2  = (const float*)d_in[10];
    const float* Wsc = (const float*)d_in[11];
    const float* gsc = (const float*)d_in[12];
    const float* bsc = (const float*)d_in[13];
    float* out = (float*)d_out;

    const int    B = 32;
    const long   M = (long)B * NPTS * KNN;   // 524288
    const long   Msc = (long)B * NPTS;       // 32768

    // -------- workspace carve-up --------
    char* wp = (char*)d_ws;
    auto carve = [&](size_t bytes) {
        char* p = wp;
        wp += (bytes + 255) & ~(size_t)255;
        return p;
    };
    __bf16* Wb0  = (__bf16*)carve((size_t)COUT * 128 * 2);
    __bf16* Wb1  = (__bf16*)carve((size_t)COUT * 128 * 2);
    __bf16* Wb2  = (__bf16*)carve((size_t)COUT * 128 * 2);
    __bf16* Wbsc = (__bf16*)carve((size_t)COUT * CIN * 2);
    float*  stats = (float*)carve(4 * 256 * sizeof(float));       // 4 sets of sum|sumsq
    float*  bnbuf = (float*)carve(8 * COUT * sizeof(float));      // s0,sh0,s1,sh1,s2,sh2,ssc,shsc
    int*    knnIdx = (int*)carve((size_t)Msc * KNN * sizeof(int));
    __bf16* fT  = (__bf16*)carve((size_t)Msc * CIN * 2);
    __bf16* scb = (__bf16*)carve((size_t)Msc * COUT * 2);
    __bf16* bufA = (__bf16*)carve((size_t)M * COUT * 2);          // x0 / y1
    __bf16* bufB = (__bf16*)carve((size_t)M * COUT * 2);          // y0 / y2

    float* s0 = bnbuf,        * sh0 = bnbuf + 128;
    float* s1 = bnbuf + 256,  * sh1 = bnbuf + 384;
    float* s2 = bnbuf + 512,  * sh2 = bnbuf + 640;
    float* ssc = bnbuf + 768, * shsc = bnbuf + 896;

    // -------- pipeline --------
    ec_zero_kernel<<<4, 256, 0, stream>>>(stats, 1024);

    ec_cvt_w_kernel<<<64, 256, 0, stream>>>(W0,  Wb0,  COUT * 128);
    ec_cvt_w_kernel<<<64, 256, 0, stream>>>(W1,  Wb1,  COUT * 128);
    ec_cvt_w_kernel<<<64, 256, 0, stream>>>(W2,  Wb2,  COUT * 128);
    ec_cvt_w_kernel<<<32, 256, 0, stream>>>(Wsc, Wbsc, COUT * CIN);

    ec_knn_kernel<<<B * 4, 256, 0, stream>>>(points, knnIdx);

    ec_build_x0_kernel<<<(unsigned)((M * COUT) / 256), 256, 0, stream>>>(features, knnIdx, bufA);

    const unsigned gemmBlocks = (unsigned)(M / 32);      // 16384
    ec_gemm_kernel<128><<<gemmBlocks, 256, 0, stream>>>(bufA, Wb0, nullptr, nullptr, bufB);
    ec_stats_kernel<<<1024, 256, 0, stream>>>(bufB, M, stats);
    ec_finalize_stats_kernel<<<1, 128, 0, stream>>>(stats, g0, b0, 1.0f / (float)M, s0, sh0);

    ec_gemm_kernel<128><<<gemmBlocks, 256, 0, stream>>>(bufB, Wb1, s0, sh0, bufA);
    ec_stats_kernel<<<1024, 256, 0, stream>>>(bufA, M, stats + 256);
    ec_finalize_stats_kernel<<<1, 128, 0, stream>>>(stats + 256, g1, b1, 1.0f / (float)M, s1, sh1);

    ec_gemm_kernel<128><<<gemmBlocks, 256, 0, stream>>>(bufA, Wb2, s1, sh1, bufB);
    ec_stats_kernel<<<1024, 256, 0, stream>>>(bufB, M, stats + 512);
    ec_finalize_stats_kernel<<<1, 128, 0, stream>>>(stats + 512, g2, b2, 1.0f / (float)M, s2, sh2);

    ec_build_ft_kernel<<<(unsigned)((Msc * CIN) / 256), 256, 0, stream>>>(features, fT);
    ec_gemm_kernel<64><<<(unsigned)(Msc / 32), 256, 0, stream>>>(fT, Wbsc, nullptr, nullptr, scb);
    ec_stats_kernel<<<256, 256, 0, stream>>>(scb, Msc, stats + 768);
    ec_finalize_stats_kernel<<<1, 128, 0, stream>>>(stats + 768, gsc, bsc, 1.0f / (float)Msc, ssc, shsc);

    ec_combine_kernel<<<(unsigned)((Msc * COUT) / 256), 256, 0, stream>>>(
        bufB, scb, s2, sh2, ssc, shsc, out);
}
